// SAC_52355651338891
// MI455X (gfx1250) — matrix operand order repaired and checked by reading, not compiled, not run on hardware
//
#include <hip/hip_runtime.h>
#include <cstdint>
#include <cstddef>

// ---------------- problem constants ----------------
static constexpr int BB   = 4;
static constexpr int LSEQ = 2048;
static constexpr int DM   = 512;
static constexpr int HN   = 8;
static constexpr int DK   = 64;          // DM / HN
static constexpr float SCALE = 0.125f;   // 1/sqrt(64)
static constexpr float NEGV  = -1e9f;

typedef unsigned short u16;
typedef __attribute__((ext_vector_type(4)))  unsigned int u32x4;
typedef __attribute__((ext_vector_type(8)))  float        v8f;
typedef __attribute__((ext_vector_type(16))) __bf16       v16bf;

// float -> bf16 (round-to-nearest-even), raw bits
static __device__ __forceinline__ u16 f2bf(float f) {
    union { float f; unsigned u; } x; x.f = f;
    unsigned r = x.u + 0x7FFFu + ((x.u >> 16) & 1u);
    return (u16)(r >> 16);
}

// Load a v16bf fragment as two 16-byte chunks: [p, p+8) and [p+off, p+off+8)
static __device__ __forceinline__ v16bf load_frag16(const u16* p, int off) {
    union { u32x4 q[2]; v16bf v; } u;
    u.q[0] = *(const u32x4*)(p);
    u.q[1] = *(const u32x4*)(p + off);
    return u.v;
}

static __device__ __forceinline__ v8f wmma_bf16(v16bf a, v16bf b, v8f c) {
    return __builtin_amdgcn_wmma_f32_16x16x32_bf16(false, a, false, b, (short)0, c, false, false);
}

static __device__ __forceinline__ float rmax16(float v) {
    #pragma unroll
    for (int o = 1; o < 16; o <<= 1) v = fmaxf(v, __shfl_xor(v, o));
    return v;
}
static __device__ __forceinline__ float rsum16(float v) {
    #pragma unroll
    for (int o = 1; o < 16; o <<= 1) v += __shfl_xor(v, o);
    return v;
}

// CDNA5 async global->LDS copy (16B per lane), tracked by ASYNCcnt.
static __device__ __forceinline__ void async_ld16(unsigned lds_addr, const u16* g) {
    asm volatile("global_load_async_to_lds_b128 %0, %1, off"
                 :: "v"(lds_addr), "v"((unsigned long long)(uintptr_t)g)
                 : "memory");
}
static __device__ __forceinline__ void wait_async0() {
    asm volatile("s_wait_asynccnt 0x0" ::: "memory");
}
static __device__ __forceinline__ unsigned lds_off(const void* p) {
    return (unsigned)(uintptr_t)p;   // low 32 bits of a shared pointer == LDS offset
}

// ---------------- f32 -> bf16 conversion ----------------
__global__ void cvt_bf16_kernel(const float* __restrict__ src, u16* __restrict__ dst, int n) {
    int i = blockIdx.x * blockDim.x + threadIdx.x;
    int stride = gridDim.x * blockDim.x;
    for (; i < n; i += stride) dst[i] = f2bf(src[i]);
}

// ---------------- WMMA GEMM:  C[M=8192, N=512] = A(bf16) @ W^T(bf16) + bias ----------------
// W is row-major [N(out), K(in)] -> its rows are B-matrix columns (K-contiguous).
// mode 0: out bf16 [M, 512]
// mode 1: out bf16 transposed per batch: [b, n, l] with m = b*2048 + l  (for V^T)
// mode 2: out f32 = acc + bias + residual  (merge + residual epilogue)
__global__ __launch_bounds__(256)
void gemm_wmma_kernel(const u16* __restrict__ A, const u16* __restrict__ W,
                      const float* __restrict__ bias, const float* __restrict__ resid,
                      u16* __restrict__ outB, float* __restrict__ outF, int mode) {
    __shared__ u16 As[2][128 * 32];   // [buf][row][k]  2 x 8KB
    __shared__ u16 Bs[2][64 * 32];    // [buf][n][k]    2 x 4KB

    const int tid   = threadIdx.x;
    const int w     = tid >> 5;
    const int lane  = tid & 31;
    const int l15   = lane & 15;
    const int lhalf = lane >> 4;
    const int wm    = w >> 1;      // 0..3
    const int wn    = w & 1;       // 0..1
    const int m0    = blockIdx.x * 128;
    const int n0    = blockIdx.y * 64;

    // per-thread staging slots: A gets 2 chunks (512 chunks / 256 thr), B gets 1
    const int rA0 = tid >> 2,          sA0 = tid & 3;
    const int rA1 = (tid + 256) >> 2,  sA1 = tid & 3;
    const int nB  = tid >> 2,          sB  = tid & 3;

    const u16* gA0 = A + (size_t)(m0 + rA0) * 512 + sA0 * 8;
    const u16* gA1 = A + (size_t)(m0 + rA1) * 512 + sA1 * 8;
    const u16* gB  = W + (size_t)(n0 + nB)  * 512 + sB  * 8;

    unsigned lA0[2], lA1[2], lB[2];
    #pragma unroll
    for (int b = 0; b < 2; ++b) {
        lA0[b] = lds_off(&As[b][rA0 * 32 + sA0 * 8]);
        lA1[b] = lds_off(&As[b][rA1 * 32 + sA1 * 8]);
        lB[b]  = lds_off(&Bs[b][nB  * 32 + sB  * 8]);
    }

    // prologue: stage tile kk=0 into buffer 0
    async_ld16(lA0[0], gA0); gA0 += 32;
    async_ld16(lA1[0], gA1); gA1 += 32;
    async_ld16(lB[0],  gB);  gB  += 32;
    wait_async0();
    __syncthreads();

    v8f acc[2][2] = {};
    int cur = 0;

    for (int kk = 0; kk < 512; kk += 32) {
        const int nb = cur ^ 1;
        if (kk + 32 < 512) {   // stream next tile while computing this one
            async_ld16(lA0[nb], gA0); gA0 += 32;
            async_ld16(lA1[nb], gA1); gA1 += 32;
            async_ld16(lB[nb],  gB);  gB  += 32;
        }

        v16bf af[2], bf[2];
        #pragma unroll
        for (int ms = 0; ms < 2; ++ms)
            af[ms] = load_frag16(&As[cur][(wm * 32 + ms * 16 + l15) * 32 + lhalf * 8], 16);
        #pragma unroll
        for (int ns = 0; ns < 2; ++ns)
            bf[ns] = load_frag16(&Bs[cur][(wn * 32 + ns * 16 + l15) * 32 + lhalf * 8], 16);
        #pragma unroll
        for (int ms = 0; ms < 2; ++ms)
            #pragma unroll
            for (int ns = 0; ns < 2; ++ns)
                acc[ms][ns] = wmma_bf16(af[ms], bf[ns], acc[ms][ns]);

        wait_async0();
        __syncthreads();
        cur = nb;
    }

    // epilogue (C layout: lane l holds col n=l&15; VGPR r holds row r + 8*(l>=16))
    #pragma unroll
    for (int ms = 0; ms < 2; ++ms) {
        #pragma unroll
        for (int ns = 0; ns < 2; ++ns) {
            int nc = n0 + wn * 32 + ns * 16 + l15;
            float bv = bias[nc];
            #pragma unroll
            for (int r = 0; r < 8; ++r) {
                int mr = m0 + wm * 32 + ms * 16 + lhalf * 8 + r;
                float v = acc[ms][ns][r] + bv;
                if (mode == 0) {
                    outB[(size_t)mr * 512 + nc] = f2bf(v);
                } else if (mode == 1) {
                    int b = mr >> 11, l = mr & 2047;
                    outB[((size_t)b * 512 + nc) * 2048 + l] = f2bf(v);
                } else {
                    size_t idx = (size_t)mr * 512 + nc;
                    outF[idx] = v + resid[idx];
                }
            }
        }
    }
}

// ---------------- Flash co-attention pass ----------------
// out[b, q, h*64+d] = softmax_k( mask_k ? NEG : (Q[q]·K[k])/8 ) @ V[k, d]
// Qm, Km: bf16 [B, L, 512] row-major.  VT: bf16 [B, 512, L] (V transposed).
__global__ __launch_bounds__(256)
void flash_dir_kernel(const u16* __restrict__ Qm, const u16* __restrict__ Km,
                      const u16* __restrict__ VT, const unsigned char* __restrict__ mask,
                      u16* __restrict__ outA) {
    __shared__ u16 Ks[2][32 * 64];    // [buf][key][d]  2 x 4KB
    __shared__ u16 Vs[2][64 * 32];    // [buf][d][key]  2 x 4KB
    __shared__ u16 Ps[8][16 * 32];    // per-wave P bounce, 8KB

    const int tid   = threadIdx.x;
    const int w     = tid >> 5;
    const int lane  = tid & 31;
    const int l15   = lane & 15;
    const int lhalf = lane >> 4;
    const int bh = blockIdx.y;
    const int b  = bh / HN;
    const int h  = bh % HN;

    const u16* Qh = Qm + (size_t)b * LSEQ * DM + h * DK;
    const u16* Kh = Km + (size_t)b * LSEQ * DM + h * DK;
    const u16* Vh = VT + (size_t)b * DM * LSEQ + (size_t)(h * DK) * LSEQ;
    const unsigned char* mb = mask + b * LSEQ;

    const int q0 = blockIdx.x * 128 + w * 16;

    // per-thread staging slots (1 x 16B chunk each for K and V^T per step)
    const int kKey = tid >> 3, kSeg = tid & 7;     // Ks: 32 rows x 8 chunks
    const int vD   = tid >> 2, vSeg = tid & 3;     // Vs: 64 rows x 4 chunks
    const u16* gK = Kh + (size_t)kKey * DM + kSeg * 8;   // advances 32*DM per step
    const u16* gV = Vh + (size_t)vD * LSEQ + vSeg * 8;   // advances 32 per step
    unsigned lK[2], lV[2];
    #pragma unroll
    for (int bi = 0; bi < 2; ++bi) {
        lK[bi] = lds_off(&Ks[bi][kKey * 64 + kSeg * 8]);
        lV[bi] = lds_off(&Vs[bi][vD * 32 + vSeg * 8]);
    }

    // A-fragments of Q (persist across the whole key loop)
    v16bf qf[2];
    {
        const u16* qrow = Qh + (size_t)(q0 + l15) * DM;
        qf[0] = load_frag16(qrow + lhalf * 8, 16);
        qf[1] = load_frag16(qrow + 32 + lhalf * 8, 16);
    }

    // prologue: stage key-chunk 0 into buffer 0
    async_ld16(lK[0], gK); gK += 32 * DM;
    async_ld16(lV[0], gV); gV += 32;
    wait_async0();
    __syncthreads();

    v8f Oa[4] = {};
    float mrow[8], lrow[8];
    #pragma unroll
    for (int r = 0; r < 8; ++r) { mrow[r] = -3.0e38f; lrow[r] = 0.0f; }

    int cur = 0;
    for (int kc = 0; kc < LSEQ; kc += 32) {
        const int nb = cur ^ 1;
        if (kc + 32 < LSEQ) {   // stream next K/V chunk while computing
            async_ld16(lK[nb], gK); gK += 32 * DM;
            async_ld16(lV[nb], gV); gV += 32;
        }

        // S = Q K^T for two 16-key subtiles (K-dim 64 = 2 wmma each)
        v8f s0 = {}, s1 = {};
        #pragma unroll
        for (int dk = 0; dk < 2; ++dk) {
            v16bf b0 = load_frag16(&Ks[cur][(l15)      * 64 + dk * 32 + lhalf * 8], 16);
            v16bf b1 = load_frag16(&Ks[cur][(16 + l15) * 64 + dk * 32 + lhalf * 8], 16);
            s0 = wmma_bf16(qf[dk], b0, s0);
            s1 = wmma_bf16(qf[dk], b1, s1);
        }

        const bool mk0 = mb[kc + l15] != 0;
        const bool mk1 = mb[kc + 16 + l15] != 0;

        float p0[8], p1[8];
        #pragma unroll
        for (int r = 0; r < 8; ++r) {
            float t0 = mk0 ? NEGV : s0[r] * SCALE;
            float t1 = mk1 ? NEGV : s1[r] * SCALE;
            float cm = rmax16(fmaxf(t0, t1));
            float mn = fmaxf(mrow[r], cm);
            float corr = __expf(mrow[r] - mn);
            float e0 = __expf(t0 - mn);
            float e1 = __expf(t1 - mn);
            float rs = rsum16(e0 + e1);
            lrow[r] = lrow[r] * corr + rs;
            mrow[r] = mn;
            Oa[0][r] *= corr; Oa[1][r] *= corr; Oa[2][r] *= corr; Oa[3][r] *= corr;
            p0[r] = e0; p1[r] = e1;
        }

        // bounce P (C-layout) through wave-private LDS into A-layout bf16.
        // Same-wave LDS ops are architecturally in-order -> compiler fence suffices.
        #pragma unroll
        for (int r = 0; r < 8; ++r) {
            int row = r + lhalf * 8;
            Ps[w][row * 32 + l15]      = f2bf(p0[r]);
            Ps[w][row * 32 + 16 + l15] = f2bf(p1[r]);
        }
        asm volatile("" ::: "memory");

        v16bf pf = load_frag16(&Ps[w][l15 * 32 + lhalf * 8], 16);
        #pragma unroll
        for (int nf = 0; nf < 4; ++nf) {
            v16bf vf = load_frag16(&Vs[cur][(nf * 16 + l15) * 32 + lhalf * 8], 16);
            Oa[nf] = wmma_bf16(pf, vf, Oa[nf]);
        }

        wait_async0();
        __syncthreads();
        cur = nb;
    }

    // finalize: divide by row sums, write merged-head layout [B, L, 512] bf16
    #pragma unroll
    for (int r = 0; r < 8; ++r) {
        float inv = 1.0f / lrow[r];
        int m = q0 + r + lhalf * 8;
        size_t base = ((size_t)b * LSEQ + m) * DM + h * DK;
        #pragma unroll
        for (int nf = 0; nf < 4; ++nf)
            outA[base + nf * 16 + l15] = f2bf(Oa[nf][r] * inv);
    }
}

// ---------------- host-side orchestration ----------------
extern "C" void kernel_launch(void* const* d_in, const int* in_sizes, int n_in,
                              void* d_out, int out_size, void* d_ws, size_t ws_size,
                              hipStream_t stream) {
    (void)in_sizes; (void)n_in; (void)out_size; (void)ws_size;

    const float* x  = (const float*)d_in[0];
    const float* y  = (const float*)d_in[1];
    const unsigned char* x_mask = (const unsigned char*)d_in[2];
    const unsigned char* y_mask = (const unsigned char*)d_in[3];
    const float* Wk  = (const float*)d_in[4];  const float* bk  = (const float*)d_in[5];
    const float* Wq  = (const float*)d_in[6];  const float* bq  = (const float*)d_in[7];
    const float* Wvx = (const float*)d_in[8];  const float* bvx = (const float*)d_in[9];
    const float* Wmx = (const float*)d_in[10]; const float* bmx = (const float*)d_in[11];
    const float* Wvy = (const float*)d_in[12]; const float* bvy = (const float*)d_in[13];
    const float* Wmy = (const float*)d_in[14]; const float* bmy = (const float*)d_in[15];

    const size_t BLD = (size_t)BB * LSEQ * DM;   // 4,194,304
    const size_t DD  = (size_t)DM * DM;          // 262,144

    u16* p = (u16*)d_ws;
    u16* Xbf  = p; p += BLD;
    u16* Ybf  = p; p += BLD;
    u16* Wkb  = p; p += DD;
    u16* Wqb  = p; p += DD;
    u16* Wvxb = p; p += DD;
    u16* Wvyb = p; p += DD;
    u16* Wmxb = p; p += DD;
    u16* Wmyb = p; p += DD;
    u16* Kbf  = p; p += BLD;
    u16* Qbf  = p; p += BLD;
    u16* VxT  = p; p += BLD;   // [B, 512, L]
    u16* VyT  = p; p += BLD;   // [B, 512, L]
    u16* Ax   = p; p += BLD;   // attedx [B, L, 512]
    u16* Ay   = p; p += BLD;   // attedy [B, L, 512]

    float* out_x = (float*)d_out;        // mergey + x
    float* out_y = out_x + BLD;          // mergex + y

    // 1) convert to bf16
    const int CT = 256, CG = 2048;
    cvt_bf16_kernel<<<CG, CT, 0, stream>>>(x, Xbf, (int)BLD);
    cvt_bf16_kernel<<<CG, CT, 0, stream>>>(y, Ybf, (int)BLD);
    cvt_bf16_kernel<<<512, CT, 0, stream>>>(Wk,  Wkb,  (int)DD);
    cvt_bf16_kernel<<<512, CT, 0, stream>>>(Wq,  Wqb,  (int)DD);
    cvt_bf16_kernel<<<512, CT, 0, stream>>>(Wvx, Wvxb, (int)DD);
    cvt_bf16_kernel<<<512, CT, 0, stream>>>(Wvy, Wvyb, (int)DD);
    cvt_bf16_kernel<<<512, CT, 0, stream>>>(Wmx, Wmxb, (int)DD);
    cvt_bf16_kernel<<<512, CT, 0, stream>>>(Wmy, Wmyb, (int)DD);

    // 2) projections: K = x@Wk^T+bk ; Q = y@Wq^T+bq ; Vx^T, Vy^T
    dim3 gg(64, 8), gb(256);
    gemm_wmma_kernel<<<gg, gb, 0, stream>>>(Xbf, Wkb,  bk,  nullptr, Kbf, nullptr, 0);
    gemm_wmma_kernel<<<gg, gb, 0, stream>>>(Ybf, Wqb,  bq,  nullptr, Qbf, nullptr, 0);
    gemm_wmma_kernel<<<gg, gb, 0, stream>>>(Xbf, Wvxb, bvx, nullptr, VxT, nullptr, 1);
    gemm_wmma_kernel<<<gg, gb, 0, stream>>>(Ybf, Wvyb, bvy, nullptr, VyT, nullptr, 1);

    // 3) flash passes (no materialized scores)
    dim3 fg(LSEQ / 128, BB * HN), fb(256);
    // x-direction: queries=q (Ly), keys=k (Lx), mask over Lx, values vx -> attedx
    flash_dir_kernel<<<fg, fb, 0, stream>>>(Qbf, Kbf, VxT, x_mask, Ax);
    // y-direction: scores transposed -> queries=k (Lx), keys=q (Ly), mask over Ly, values vy -> attedy
    flash_dir_kernel<<<fg, fb, 0, stream>>>(Kbf, Qbf, VyT, y_mask, Ay);

    // 4) merge + residual (f32 epilogue straight to d_out)
    gemm_wmma_kernel<<<gg, gb, 0, stream>>>(Ay, Wmyb, bmy, x, nullptr, out_x, 2);
    gemm_wmma_kernel<<<gg, gb, 0, stream>>>(Ax, Wmxb, bmx, y, nullptr, out_y, 2);
}
